// Network_75084618269290
// MI455X (gfx1250) — matrix operand order repaired and checked by reading, not compiled, not run on hardware
//
#include <hip/hip_runtime.h>
#include <hip/hip_bf16.h>

// ---------------------------------------------------------------------------
// Problem constants (from reference setup_inputs)
// ---------------------------------------------------------------------------
#define BB      8
#define INP     257
#define TT      2000
#define CC      512
#define KK      9
#define DD      4

#define CPAD_IN   288   // 257 padded to multiple of 32
#define CBLK_IN   9     // 288/32
#define CBLK_MID  16    // 512/32
#define MBLK_C    32    // 512/16
#define MOUT_PAD  272   // 257 padded to multiple of 16
#define MBLK_OUT  17    // 272/16

#define TILE_T    32    // t-columns per workgroup
#define HALO      4     // K/2
#define ROWS      (TILE_T + 2 * HALO)   // 40 staged rows
#define LDS_PAD   8     // elements of row padding (16B) -> bank-conflict-free

#if __has_builtin(__builtin_amdgcn_sched_barrier)
#define SCHED_FENCE() __builtin_amdgcn_sched_barrier(0)
#else
#define SCHED_FENCE()
#endif

typedef __attribute__((ext_vector_type(16))) __bf16 v16bf;
typedef __attribute__((ext_vector_type(8)))  float  v8f;

union Frag32B { uint4 u[2]; v16bf v; };

// ---------------------------------------------------------------------------
// BN prep: scale = gamma*rsqrt(var+eps), bias = beta - mean*scale; zero pad.
// ---------------------------------------------------------------------------
__global__ void bn_prep_kernel(const float* __restrict__ g, const float* __restrict__ be,
                               const float* __restrict__ mn, const float* __restrict__ vr,
                               float* __restrict__ scale, float* __restrict__ bias,
                               int n, int npad)
{
    int i = blockIdx.x * 256 + threadIdx.x;
    if (i >= npad) return;
    if (i < n) {
        float inv = g[i] * rsqrtf(vr[i] + 1e-5f);
        scale[i] = inv;
        bias[i]  = be[i] - mn[i] * inv;
    } else {
        scale[i] = 0.f;
        bias[i]  = 0.f;
    }
}

// ---------------------------------------------------------------------------
// Weight pack: clamp(w,-1,1) -> bf16, laid out in WMMA A-fragment order.
// Block = (tap, cblk, mb): 32 lanes x 16 bf16 contiguous per lane.
// A 16x32 order used here (contiguous k, B matches): M = lane%16;
//   lanes 0-15 : K = e       (K in {0..15})
//   lanes16-31 : K = 16 + e  (K in {16..31})
// ---------------------------------------------------------------------------
__global__ void pack_weights_kernel(const float* __restrict__ w, __hip_bfloat16* __restrict__ dst,
                                    int Cout, int Cin, int CBLKS, int MBLKS, long total)
{
    long idx = (long)blockIdx.x * 256 + threadIdx.x;
    if (idx >= total) return;
    int e    = (int)(idx & 15);
    int lane = (int)((idx >> 4) & 31);
    long blk = idx >> 9;
    int mb   = (int)(blk % MBLKS);  blk /= MBLKS;
    int cb   = (int)(blk % CBLKS);
    int tap  = (int)(blk / CBLKS);

    int m     = mb * 16 + (lane & 15);
    int kbase = (lane < 16) ? 0 : 16;
    int koff  = e;
    int cin   = cb * 32 + kbase + koff;

    float val = 0.f;
    if (m < Cout && cin < Cin) {
        val = w[((size_t)m * Cin + cin) * KK + tap];
        val = fminf(1.f, fmaxf(-1.f, val));
    }
    dst[idx] = __float2bfloat16(val);
}

// ---------------------------------------------------------------------------
// Input prep: noisy [B][INP][T] f32 -> act [B][T][CPAD_IN] bf16 (zero-padded).
// ---------------------------------------------------------------------------
__global__ void prep_input_kernel(const float* __restrict__ noisy, __hip_bfloat16* __restrict__ act)
{
    long idx = (long)blockIdx.x * 256 + threadIdx.x;
    const long total = (long)BB * TT * CPAD_IN;
    if (idx >= total) return;
    int c = (int)(idx % CPAD_IN);
    long bt = idx / CPAD_IN;
    int t = (int)(bt % TT);
    int b = (int)(bt / TT);
    float v = (c < INP) ? noisy[((size_t)b * INP + c) * TT + t] : 0.f;
    act[idx] = __float2bfloat16(v);
}

// ---------------------------------------------------------------------------
// Fused conv1d(SAME,K=9) + BN + ReLU via v_wmma_f32_16x16x32_bf16.
//
// - Activation tile (32 t + 8 halo rows, zero-filled OOB) staged once into LDS;
//   B-fragments come from ds_load_b128 (bank-conflict-free +16B row padding).
// - K-step loop (tap,cb flattened; weight address purely linear) is manually
//   software-pipelined: double-buffered A/B fragments + sched_barrier so loads
//   for step s+1 are in flight while the matrix pipe runs step s.
// - WG = 8 waves: 4-way M split x 2-way N split; tile = Cout_pad x 32 t.
// ---------------------------------------------------------------------------
template<int MB_PER, int MBLKS, int CINPAD, int CBLKS>
__global__ __launch_bounds__(256)
void conv_bn_relu_wmma(const __hip_bfloat16* __restrict__ actIn,
                       const __hip_bfloat16* __restrict__ Wp,
                       const float* __restrict__ scale, const float* __restrict__ bias,
                       __hip_bfloat16* __restrict__ actOut, int CoutPad,
                       float* __restrict__ outF32, int CoutValid, int T)
{
    constexpr int STRIDE_E = CINPAD + LDS_PAD;           // elements per LDS row
    constexpr int KSTEPS   = KK * CBLKS;
    __shared__ __align__(16) __hip_bfloat16 smem[ROWS * STRIDE_E];

    const int lane  = threadIdx.x & 31;
    const int wave  = threadIdx.x >> 5;
    const int mq    = wave & 3;        // M quarter
    const int nsub  = wave >> 2;       // N sub-tile (0,1)
    const int b     = blockIdx.y;
    const int tbase = blockIdx.x * TILE_T;
    const int tcol  = tbase + nsub * 16 + (lane & 15);   // this lane's output t
    const int cHalf = (lane >> 4) << 4;                  // B-frag c offset: 0 or 16

    // ---- stage activation tile (rows tbase-4 .. tbase+35) into LDS ----
    {
        constexpr int CHUNKS_PER_ROW = CINPAD / 8;       // uint4 chunks
        constexpr int TOTAL_CHUNKS   = ROWS * CHUNKS_PER_ROW;
        const __hip_bfloat16* actB0 = actIn + (size_t)b * T * CINPAD;
        for (int idx = threadIdx.x; idx < TOTAL_CHUNKS; idx += 256) {
            const int r   = idx / CHUNKS_PER_ROW;
            const int col = (idx % CHUNKS_PER_ROW) * 8;  // element offset in row
            const int tsrc = tbase - HALO + r;
            uint4 val = uint4{0u, 0u, 0u, 0u};
            if (tsrc >= 0 && tsrc < T)
                val = *reinterpret_cast<const uint4*>(actB0 + (size_t)tsrc * CINPAD + col);
            *reinterpret_cast<uint4*>(&smem[r * STRIDE_E + col]) = val;
        }
    }
    __syncthreads();

    const v8f vzero = {0.f, 0.f, 0.f, 0.f, 0.f, 0.f, 0.f, 0.f};
    v8f acc[MB_PER];
#pragma unroll
    for (int i = 0; i < MB_PER; ++i) acc[i] = vzero;

    // LDS row for this lane at tap=0 (row index = local t + tap, halo built in)
    const __hip_bfloat16* ldsRow0 = smem + (nsub * 16 + (lane & 15)) * STRIDE_E + cHalf;
    // This wave's weight base; step s adds s*MBLKS*512 elements (pack order is
    // (tap,cb)-row-major so the K-step weight walk is purely linear).
    const __hip_bfloat16* wlane = Wp + ((size_t)mq * MB_PER) * 512 + lane * 16;
    constexpr size_t WSTEP = (size_t)MBLKS * 512;

    auto loadA = [&](Frag32B (&af)[MB_PER], int s) {
        const __hip_bfloat16* wp = wlane + (size_t)s * WSTEP;
#pragma unroll
        for (int i = 0; i < MB_PER; ++i) {
            if (MB_PER * 4 == MBLKS || (mq * MB_PER + i < MBLKS)) {
                const uint4* ap = reinterpret_cast<const uint4*>(wp + (size_t)i * 512);
                af[i].u[0] = ap[0];
                af[i].u[1] = ap[1];
            }
        }
    };
    auto loadB = [&](Frag32B& bf, int s) {
        const int tap = s / CBLKS;
        const int cb  = s - tap * CBLKS;
        const uint4* bp = reinterpret_cast<const uint4*>(ldsRow0 + tap * STRIDE_E + cb * 32);
        bf.u[0] = bp[0];
        bf.u[1] = bp[1];
    };
    auto mmaStep = [&](Frag32B (&af)[MB_PER], Frag32B& bf) {
#pragma unroll
        for (int i = 0; i < MB_PER; ++i) {
            if (MB_PER * 4 == MBLKS || (mq * MB_PER + i < MBLKS)) {
                acc[i] = __builtin_amdgcn_wmma_f32_16x16x32_bf16(
                    false, af[i].v, false, bf.v, (short)0, acc[i], false, false);
            }
        }
    };

    // ---- software-pipelined K loop: load (s+1) while MMA-ing (s) ----
    Frag32B afX[MB_PER], afY[MB_PER];
    Frag32B bfX, bfY;
    loadA(afX, 0);
    loadB(bfX, 0);

    int s = 0;
    for (; s + 2 <= KSTEPS; s += 2) {
        loadA(afY, s + 1);
        loadB(bfY, s + 1);
        __builtin_prefetch(wlane + (size_t)((s + 3 < KSTEPS) ? s + 3 : KSTEPS - 1) * WSTEP, 0, 1);
        SCHED_FENCE();
        mmaStep(afX, bfX);
        SCHED_FENCE();

        const int snext = (s + 2 < KSTEPS) ? (s + 2) : (KSTEPS - 1);
        loadA(afX, snext);
        loadB(bfX, snext);
        SCHED_FENCE();
        mmaStep(afY, bfY);
        SCHED_FENCE();
    }
    if (s < KSTEPS) {   // odd KSTEPS tail: afX/bfX already hold step KSTEPS-1
        mmaStep(afX, bfX);
    }

    // ---- epilogue: BN + ReLU. D layout: VGPR r -> M=r (lanes 0-15) / M=8+r ----
    const bool tIn   = (tcol < T);
    const int  half8 = (lane >> 4) << 3;
#pragma unroll
    for (int i = 0; i < MB_PER; ++i) {
        const int mb = mq * MB_PER + i;
        if (mb >= MBLKS) continue;
        const int cbase = mb * 16 + half8;
#pragma unroll
        for (int r = 0; r < 8; ++r) {
            const int c = cbase + r;
            float v = fmaxf(acc[i][r] * scale[c] + bias[c], 0.0f);
            if (outF32) {
                if (tIn && c < CoutValid)
                    outF32[((size_t)b * CoutValid + c) * T + tcol] = v;   // [B][257][T] f32
            } else {
                if (tIn)
                    actOut[((size_t)b * T + tcol) * CoutPad + c] = __float2bfloat16(v);
            }
        }
    }
}

// ---------------------------------------------------------------------------
// Host launcher
// ---------------------------------------------------------------------------
static inline int cdiv_i(long a, long b) { return (int)((a + b - 1) / b); }

extern "C" void kernel_launch(void* const* d_in, const int* in_sizes, int n_in,
                              void* d_out, int out_size, void* d_ws, size_t ws_size,
                              hipStream_t stream)
{
    const float* noisy = (const float*)d_in[0];
    const float* w_in  = (const float*)d_in[1];
    const float* w_mid = (const float*)d_in[2];
    const float* w_out = (const float*)d_in[3];
    const float* g_in  = (const float*)d_in[4];
    const float* b_in  = (const float*)d_in[5];
    const float* m_in  = (const float*)d_in[6];
    const float* v_in  = (const float*)d_in[7];
    const float* g_mid = (const float*)d_in[8];
    const float* b_mid = (const float*)d_in[9];
    const float* m_mid = (const float*)d_in[10];
    const float* v_mid = (const float*)d_in[11];
    const float* g_out = (const float*)d_in[12];
    const float* b_out = (const float*)d_in[13];
    const float* m_out = (const float*)d_in[14];
    const float* v_out = (const float*)d_in[15];

    // ----- workspace layout (all sizes are multiples of 256 bytes) -----
    constexpr size_t SZ_ACT0   = (size_t)BB * TT * CPAD_IN * 2;                 //  9,216,000
    constexpr size_t SZ_ACT    = (size_t)BB * TT * CC * 2;                      // 16,384,000
    constexpr size_t SZ_WPIN   = (size_t)KK * CBLK_IN  * MBLK_C  * 512 * 2;     //  2,654,208
    constexpr size_t SZ_WPMID1 = (size_t)KK * CBLK_MID * MBLK_C  * 512 * 2;     //  4,718,592
    constexpr size_t SZ_WPOUT  = (size_t)KK * CBLK_MID * MBLK_OUT * 512 * 2;    //  2,506,752

    char* ws = (char*)d_ws;
    __hip_bfloat16* act0  = (__hip_bfloat16*)ws;                       ws += SZ_ACT0;
    __hip_bfloat16* actA  = (__hip_bfloat16*)ws;                       ws += SZ_ACT;
    __hip_bfloat16* actB  = (__hip_bfloat16*)ws;                       ws += SZ_ACT;
    __hip_bfloat16* wpIn  = (__hip_bfloat16*)ws;                       ws += SZ_WPIN;
    __hip_bfloat16* wpMid = (__hip_bfloat16*)ws;                       ws += (size_t)DD * SZ_WPMID1;
    __hip_bfloat16* wpOut = (__hip_bfloat16*)ws;                       ws += SZ_WPOUT;
    float* scIn  = (float*)ws;  ws += 512 * sizeof(float);
    float* biIn  = (float*)ws;  ws += 512 * sizeof(float);
    float* scMid = (float*)ws;  ws += (size_t)DD * 512 * sizeof(float);
    float* biMid = (float*)ws;  ws += (size_t)DD * 512 * sizeof(float);
    float* scOut = (float*)ws;  ws += MOUT_PAD * sizeof(float);
    float* biOut = (float*)ws;  ws += MOUT_PAD * sizeof(float);

    // ----- BN folding -----
    bn_prep_kernel<<<cdiv_i(512, 256), 256, 0, stream>>>(g_in, b_in, m_in, v_in, scIn, biIn, 512, 512);
    bn_prep_kernel<<<cdiv_i(DD * 512, 256), 256, 0, stream>>>(g_mid, b_mid, m_mid, v_mid, scMid, biMid, DD * 512, DD * 512);
    bn_prep_kernel<<<cdiv_i(MOUT_PAD, 256), 256, 0, stream>>>(g_out, b_out, m_out, v_out, scOut, biOut, INP, MOUT_PAD);

    // ----- weight packing (clip + bf16 + WMMA A-fragment order) -----
    {
        long nIn = (long)KK * CBLK_IN * MBLK_C * 512;
        pack_weights_kernel<<<cdiv_i(nIn, 256), 256, 0, stream>>>(w_in, wpIn, CC, INP, CBLK_IN, MBLK_C, nIn);
        long nMid = (long)KK * CBLK_MID * MBLK_C * 512;
        for (int l = 0; l < DD; ++l) {
            pack_weights_kernel<<<cdiv_i(nMid, 256), 256, 0, stream>>>(
                w_mid + (size_t)l * CC * CC * KK, wpMid + (size_t)l * nMid, CC, CC, CBLK_MID, MBLK_C, nMid);
        }
        long nOut = (long)KK * CBLK_MID * MBLK_OUT * 512;
        pack_weights_kernel<<<cdiv_i(nOut, 256), 256, 0, stream>>>(w_out, wpOut, INP, CC, CBLK_MID, MBLK_OUT, nOut);
    }

    // ----- input transpose/pad to time-major bf16 -----
    prep_input_kernel<<<cdiv_i((long)BB * TT * CPAD_IN, 256), 256, 0, stream>>>(noisy, act0);

    // ----- conv stack -----
    dim3 grid(cdiv_i(TT, TILE_T), BB);
    const long nMid = (long)KK * CBLK_MID * MBLK_C * 512;

    conv_bn_relu_wmma<8, MBLK_C, CPAD_IN, CBLK_IN><<<grid, 256, 0, stream>>>(
        act0, wpIn, scIn, biIn, actA, CC, nullptr, 0, TT);

    conv_bn_relu_wmma<8, MBLK_C, CC, CBLK_MID><<<grid, 256, 0, stream>>>(
        actA, wpMid + 0 * nMid, scMid + 0 * 512, biMid + 0 * 512, actB, CC, nullptr, 0, TT);
    conv_bn_relu_wmma<8, MBLK_C, CC, CBLK_MID><<<grid, 256, 0, stream>>>(
        actB, wpMid + 1 * nMid, scMid + 1 * 512, biMid + 1 * 512, actA, CC, nullptr, 0, TT);
    conv_bn_relu_wmma<8, MBLK_C, CC, CBLK_MID><<<grid, 256, 0, stream>>>(
        actA, wpMid + 2 * nMid, scMid + 2 * 512, biMid + 2 * 512, actB, CC, nullptr, 0, TT);
    conv_bn_relu_wmma<8, MBLK_C, CC, CBLK_MID><<<grid, 256, 0, stream>>>(
        actB, wpMid + 3 * nMid, scMid + 3 * 512, biMid + 3 * 512, actA, CC, nullptr, 0, TT);

    conv_bn_relu_wmma<5, MBLK_OUT, CC, CBLK_MID><<<grid, 256, 0, stream>>>(
        actA, wpOut, scOut, biOut, nullptr, 0, (float*)d_out, INP, TT);
}
